// TrafficModel_44547400794205
// MI455X (gfx1250) — matrix operand order repaired and checked by reading, not compiled, not run on hardware
//
#include <hip/hip_runtime.h>
#include <stdint.h>

typedef __attribute__((ext_vector_type(16))) _Float16 v16h;
typedef __attribute__((ext_vector_type(8)))  _Float16 v8h;
typedef __attribute__((ext_vector_type(4)))  _Float16 v4h;
typedef __attribute__((ext_vector_type(8)))  float    v8f;
typedef __attribute__((ext_vector_type(4)))  float    v4f;
typedef __attribute__((ext_vector_type(4)))  unsigned int v4u;
typedef __attribute__((ext_vector_type(4)))  int      v4i;
typedef __attribute__((ext_vector_type(8)))  int      v8i;

#define N_ROUTES 512
#define N_TIMES  4096
#define N_MOVES  16384
#define N_NODES  2048
#define N_CDF    8192
#define ALPHA    0.15f

// ---- CDNA5 Tensor Data Mover availability ----
#if defined(__has_builtin)
# if __has_builtin(__builtin_amdgcn_tensor_load_to_lds)
#  define USE_TDM 1
# endif
#endif
#ifndef USE_TDM
# define USE_TDM 0
#endif
#if __has_include(<hip/amd_detail/amd_gfx1250_TDM.h>)
# define TDM_6ARG 1
#else
# define TDM_6ARG 0
#endif

#if USE_TDM
// 2-D TDM tile load: tile = rows_tile x 32 halfwords from a row-major [rows_tensor x 512] f16
// tensor; LDS rows padded by 4 DWORDs every 16 DWORDs -> 40-halfword LDS row stride.
static __device__ __forceinline__ void tdm_load_2d(const _Float16* g, unsigned lds_off,
                                                   unsigned rows_tensor, unsigned rows_tile) {
    unsigned long long ga = (unsigned long long)(uintptr_t)g;
    // D# group 0: count=1 | lds_addr | global_addr[56:0] | type=2
    v4u g0 = { 1u, lds_off, (unsigned)ga,
               (unsigned)((ga >> 32) & 0x1FFFFFFu) | 0x80000000u };
    // D# group 1: data_size=2B, pad_enable, pad_interval=16dw(code 3), pad_amount=4dw(code 3),
    // tensor_dim0=512, tensor_dim1=rows_tensor, tile_dim0=32, tile_dim1=rows_tile,
    // tensor_dim0_stride=512
    v8i g1 = { (int)(0x10000u | (1u << 20) | (3u << 22) | (3u << 25)),
               (int)(512u << 16),
               (int)(rows_tensor << 16),
               (int)(32u << 16),
               (int)rows_tile,
               512, 0, 0 };
    v4i z4 = { 0, 0, 0, 0 };
#if TDM_6ARG
    v8i z8 = { 0, 0, 0, 0, 0, 0, 0, 0 };
    __builtin_amdgcn_tensor_load_to_lds(g0, g1, z4, z4, z8, 0);
#else
    __builtin_amdgcn_tensor_load_to_lds(g0, g1, z4, z4, 0);
#endif
}
#endif

// ---------------- elementwise kernels ----------------

// x relu -> f32 output (exact)
__global__ __launch_bounds__(256) void k_relu(const float* __restrict__ xr,
                                              float* __restrict__ xo, int n4) {
    int i = blockIdx.x * 256 + threadIdx.x;
    if (i >= n4) return;
    int j = i * 4;
    v4f v = *(const v4f*)(xr + j);
    v4f r = { fmaxf(v.x, 0.f), fmaxf(v.y, 0.f), fmaxf(v.z, 0.f), fmaxf(v.w, 0.f) };
    *(v4f*)(xo + j) = r;
}

// relu + f16 + transpose: Xt[t][r] = relu(x_raw[r][t]); 32x32 tiles via LDS
__global__ __launch_bounds__(256) void k_xt(const float* __restrict__ xr,
                                            _Float16* __restrict__ Xt) {
    __shared__ _Float16 tile[32][40];
    int t0 = blockIdx.x * 32, r0 = blockIdx.y * 32;
    int c = threadIdx.x & 31, rr = threadIdx.x >> 5;
    #pragma unroll
    for (int i = 0; i < 4; ++i) {
        int r = rr + i * 8;
        float v = xr[(size_t)(r0 + r) * N_TIMES + t0 + c];
        tile[c][r] = (_Float16)fmaxf(v, 0.f);
    }
    __syncthreads();
    #pragma unroll
    for (int i = 0; i < 4; ++i) {
        int tt = rr + i * 8;
        Xt[(size_t)(t0 + tt) * N_ROUTES + r0 + c] = tile[tt][c];
    }
}

// f32 -> f16 (4 elts/thread)
__global__ __launch_bounds__(256) void k_f32_to_f16(const float* __restrict__ in,
                                                    _Float16* __restrict__ out, int n4) {
    int i = blockIdx.x * 256 + threadIdx.x;
    if (i >= n4) return;
    int j = i * 4;
    v4f v = *(const v4f*)(in + j);
    v4h h = { (_Float16)v.x, (_Float16)v.y, (_Float16)v.z, (_Float16)v.w };
    *(v4h*)(out + j) = h;
}

__global__ __launch_bounds__(256) void k_zero(float* __restrict__ p, int n4) {
    int i = blockIdx.x * 256 + threadIdx.x;
    if (i >= n4) return;
    v4f z = { 0.f, 0.f, 0.f, 0.f };
    *(v4f*)(p + i * 4) = z;
}

// SA[node][r] += A[m][r] for movement_node[m]==node  (S@A reduction, f32 atomics)
__global__ __launch_bounds__(256) void k_sa_fill(const float* __restrict__ A,
                                                 const int* __restrict__ mnode,
                                                 float* __restrict__ SA) {
    int idx = blockIdx.x * 256 + threadIdx.x;       // 16384 * 128
    int m  = idx >> 7;
    int r4 = (idx & 127) * 4;
    v4f v = *(const v4f*)(A + (size_t)m * N_ROUTES + r4);
    int node = mnode[m];
    float* p = SA + (size_t)node * N_ROUTES + r4;
    if (v.x != 0.f) atomicAdd(p + 0, v.x);
    if (v.y != 0.f) atomicAdd(p + 1, v.y);
    if (v.z != 0.f) atomicAdd(p + 2, v.z);
    if (v.w != 0.f) atomicAdd(p + 3, v.w);
}

// ---------------- WMMA GEMM: C[M x 4096] = Af16[M x 512] * Xt[4096 x 512]^T --------------
// block = 8 waves, tile BM=128, BN=64, BK=32; wave owns a 16x64 strip (4 wmma tiles).
// TDM path: double-buffered LDS tiles, the Tensor Data Mover streams tile i+1 while
// the 8 waves run WMMA on tile i (one barrier per k-step).

#define AS_STR 40   // 32 + 8 halfword pad (80B row -> 20-bank stride, conflict-free)
#define BS_STR 40
#define KSTEPS (N_ROUTES / 32)

__global__ __launch_bounds__(256) void k_gemm_f16(const _Float16* __restrict__ Af,
                                                  const _Float16* __restrict__ Xt,
                                                  float* __restrict__ C) {
#if USE_TDM
    __shared__ _Float16 As[2][128 * AS_STR];
    __shared__ _Float16 Bt[2][64 * BS_STR];
#else
    __shared__ _Float16 As[1][128 * AS_STR];
    __shared__ _Float16 Bt[1][64 * BS_STR];
#endif

    const int tid  = threadIdx.x;
    const int wave = tid >> 5, lane = tid & 31;
    const int g = lane >> 4, ln = lane & 15;
    const int m0 = blockIdx.y * 128, n0 = blockIdx.x * 64;

    v8f acc0 = {}, acc1 = {}, acc2 = {}, acc3 = {};

#if USE_TDM
    const unsigned ldsA0 = (unsigned)(uintptr_t)(&As[0][0]);
    const unsigned ldsA1 = (unsigned)(uintptr_t)(&As[1][0]);
    const unsigned ldsB0 = (unsigned)(uintptr_t)(&Bt[0][0]);
    const unsigned ldsB1 = (unsigned)(uintptr_t)(&Bt[1][0]);
    const unsigned Mrows = gridDim.y * 128u;
    const _Float16* Abase = Af + (size_t)m0 * N_ROUTES;
    const _Float16* Bbase = Xt + (size_t)n0 * N_ROUTES;

    if (tid < 32) {   // prologue: TDM fills buffer 0
        tdm_load_2d(Abase, ldsA0, Mrows, 128u);
        tdm_load_2d(Bbase, ldsB0, (unsigned)N_TIMES, 64u);
    }

    #pragma unroll 1
    for (int it = 0; it < KSTEPS; ++it) {
        const int cur = it & 1;
        if (tid < 32) __builtin_amdgcn_s_wait_tensorcnt(0);   // tile `it` landed
        __syncthreads();                                      // publish; prior reads drained
        if (tid < 32 && (it + 1) < KSTEPS) {                  // stream tile it+1 during compute
            const int kn = (it + 1) * 32;
            tdm_load_2d(Abase + kn, cur ? ldsA0 : ldsA1, Mrows, 128u);
            tdm_load_2d(Bbase + kn, cur ? ldsB0 : ldsB1, (unsigned)N_TIMES, 64u);
        }
#else
    const int am = tid >> 1, asub = (tid & 1) * 16;   // A stage: 32B per thread
    const int bn = tid >> 2, bsub = (tid & 3) * 8;    // B stage: 16B per thread
    #pragma unroll 1
    for (int it = 0; it < KSTEPS; ++it) {
        const int cur = 0;
        const int k0 = it * 32;
        v16h av = *(const v16h*)(Af + (size_t)(m0 + am) * N_ROUTES + k0 + asub);
        *(v16h*)(&As[0][am * AS_STR + asub]) = av;
        v8h bv = *(const v8h*)(Xt + (size_t)(n0 + bn) * N_ROUTES + k0 + bsub);
        *(v8h*)(&Bt[0][bn * BS_STR + bsub]) = bv;
        __syncthreads();
#endif

        // A fragment: lane ln = row; halfwords 0..7 = K 8g..8g+7, 8..15 = K 16+8g..23+8g
        const _Float16* ap = &As[cur][(wave * 16 + ln) * AS_STR];
        v8h alo = *(const v8h*)(ap + 8 * g);
        v8h ahi = *(const v8h*)(ap + 16 + 8 * g);
        v16h a = __builtin_shufflevector(alo, ahi, 0,1,2,3,4,5,6,7,8,9,10,11,12,13,14,15);

        // B fragments: lane ln = column; halfwords 0..7 = K 8g..8g+7, 8..15 = K 16+8g..23+8g
        const _Float16* bp = &Bt[cur][ln * BS_STR];
        auto ldb = [&](const _Float16* p) { v8h lo = *(const v8h*)(p + 8 * g); v8h hi = *(const v8h*)(p + 16 + 8 * g); return __builtin_shufflevector(lo, hi, 0,1,2,3,4,5,6,7,8,9,10,11,12,13,14,15); };
        v16h b0 = ldb(bp);
        v16h b1 = ldb(bp + 16 * BS_STR);
        v16h b2 = ldb(bp + 32 * BS_STR);
        v16h b3 = ldb(bp + 48 * BS_STR);

        acc0 = __builtin_amdgcn_wmma_f32_16x16x32_f16(false, a, false, b0, (short)0, acc0, false, false);
        acc1 = __builtin_amdgcn_wmma_f32_16x16x32_f16(false, a, false, b1, (short)0, acc1, false, false);
        acc2 = __builtin_amdgcn_wmma_f32_16x16x32_f16(false, a, false, b2, (short)0, acc2, false, false);
        acc3 = __builtin_amdgcn_wmma_f32_16x16x32_f16(false, a, false, b3, (short)0, acc3, false, false);

#if !USE_TDM
        __syncthreads();   // single-buffer fallback needs a tail barrier
#endif
    }

    // D layout: VGPR r -> row r + 8*g, lane%16 -> col
    const int row0 = m0 + wave * 16 + 8 * g;
    float* cp = C + (size_t)row0 * N_TIMES + n0 + ln;
    #pragma unroll
    for (int r = 0; r < 8; ++r) {
        cp[(size_t)r * N_TIMES +  0] = acc0[r];
        cp[(size_t)r * N_TIMES + 16] = acc1[r];
        cp[(size_t)r * N_TIMES + 32] = acc2[r];
        cp[(size_t)r * N_TIMES + 48] = acc3[r];
    }
}

// ---------------- BPR delay + column reduction ----------------
// y[t] = sum_c t_free[c]*radio[c]*(1 + ALPHA * max(flow/cap, 1e-6)^4)
__global__ __launch_bounds__(256) void k_bpr(const float* __restrict__ nf,
                                             const int* __restrict__ cdf_node,
                                             const float* __restrict__ tf,
                                             const float* __restrict__ cap,
                                             const float* __restrict__ radio,
                                             float* __restrict__ y) {
    int t  = blockIdx.x * 256 + threadIdx.x;
    int c0 = blockIdx.y * 1024;
    float acc = 0.f;
    #pragma unroll 4
    for (int c = c0; c < c0 + 1024; ++c) {
        int node = cdf_node[c];                       // uniform -> scalar loads
        float flow  = nf[(size_t)node * N_TIMES + t];
        float ratio = fmaxf(flow / cap[c], 1e-6f);
        float r2 = ratio * ratio;
        acc += tf[c] * radio[c] * (1.0f + ALPHA * r2 * r2);
    }
    atomicAdd(&y[t], acc);
}

// ---------------- launch ----------------

extern "C" void kernel_launch(void* const* d_in, const int* in_sizes, int n_in,
                              void* d_out, int out_size, void* d_ws, size_t ws_size,
                              hipStream_t stream) {
    (void)in_sizes; (void)n_in; (void)out_size; (void)ws_size;
    const float* x_raw  = (const float*)d_in[0];
    const float* A      = (const float*)d_in[1];
    const float* t_free = (const float*)d_in[2];
    const float* cap    = (const float*)d_in[3];
    const float* radio  = (const float*)d_in[4];
    const int*   mnode  = (const int*)d_in[5];
    const int*   cdf    = (const int*)d_in[6];

    float* out   = (float*)d_out;
    float* out_x = out;
    float* out_c = out + (size_t)N_ROUTES * N_TIMES;
    float* out_y = out_c + (size_t)N_MOVES * N_TIMES;

    char* ws = (char*)d_ws;
    _Float16* Xt   = (_Float16*)ws;                                            // 4 MB
    _Float16* A16  = (_Float16*)(ws + (size_t)N_ROUTES * N_TIMES * 2);         // 16 MB
    float*    SA   = (float*)(ws + (size_t)N_ROUTES * N_TIMES * 2
                                 + (size_t)N_MOVES * N_ROUTES * 2);            // 4 MB
    _Float16* SA16 = (_Float16*)((char*)SA + (size_t)N_NODES * N_ROUTES * 4);  // 2 MB
    float*    NF   = (float*)((char*)SA16 + (size_t)N_NODES * N_ROUTES * 2);   // 32 MB

    // x = relu(x_raw) (f32 output) and Xt = relu(x)^T in f16
    k_relu<<<(N_ROUTES * N_TIMES / 4 + 255) / 256, 256, 0, stream>>>(
        x_raw, out_x, N_ROUTES * N_TIMES / 4);
    dim3 gt(N_TIMES / 32, N_ROUTES / 32);
    k_xt<<<gt, 256, 0, stream>>>(x_raw, Xt);

    // A -> f16
    k_f32_to_f16<<<(N_MOVES * N_ROUTES / 4 + 255) / 256, 256, 0, stream>>>(
        A, A16, N_MOVES * N_ROUTES / 4);

    // SA = S@A (segment-sum of A rows by node), then f16
    k_zero<<<(N_NODES * N_ROUTES / 4 + 255) / 256, 256, 0, stream>>>(SA, N_NODES * N_ROUTES / 4);
    k_zero<<<4, 256, 0, stream>>>(out_y, N_TIMES / 4);
    k_sa_fill<<<(N_MOVES * (N_ROUTES / 4) + 255) / 256, 256, 0, stream>>>(A, mnode, SA);
    k_f32_to_f16<<<(N_NODES * N_ROUTES / 4 + 255) / 256, 256, 0, stream>>>(
        SA, SA16, N_NODES * N_ROUTES / 4);

    // c_pred = A @ x   (16384 x 4096)
    dim3 gc(N_TIMES / 64, N_MOVES / 128);
    k_gemm_f16<<<gc, 256, 0, stream>>>(A16, Xt, out_c);
    // node_flow = (S@A) @ x   (2048 x 4096)
    dim3 gn(N_TIMES / 64, N_NODES / 128);
    k_gemm_f16<<<gn, 256, 0, stream>>>(SA16, Xt, NF);

    // y reduction over the 8192 cdf rows
    dim3 gy(N_TIMES / 256, N_CDF / 1024);
    k_bpr<<<gy, 256, 0, stream>>>(NF, cdf, t_free, cap, radio, out_y);
}
